// MultiheadGraphAttention_88888643158208
// MI455X (gfx1250) — matrix operand order repaired and checked by reading, not compiled, not run on hardware
//
#include <hip/hip_runtime.h>
#include <hip/hip_bf16.h>
#include <cstddef>

// ---------------------------------------------------------------------------
// MultiheadGraphAttention (DGCNN-style) for MI455X / gfx1250, wave32 + WMMA.
// B=4, CIN=64, N=4096, H=4, CH=64, K=20.  All GEMMs via v_wmma_f32_16x16x32_f16.
// Weight matrices are pre-swizzled into WMMA-B fragment-major LDS layout so
// every fragment load is a contiguous 32B ds_load_b128 pair (no u16 gathers).
// ---------------------------------------------------------------------------

#define BATCH  4
#define CIN    64
#define NPTS   4096
#define HEADS  4
#define CH     64
#define KNN    20
#define NEG_SLOPE 0.2f

typedef _Float16 half_t;
typedef __attribute__((ext_vector_type(16))) _Float16 v16h;
typedef __attribute__((ext_vector_type(8)))  _Float16 v8h;
typedef __attribute__((ext_vector_type(8)))  float    v8f;
typedef __attribute__((ext_vector_type(4)))  float    v4f;

// ---- WMMA helpers (CDNA5 wave32 layouts, CDNA5 ISA) --------------

__device__ __forceinline__ v8f wmma_f16(v16h a, v16h b, v8f c) {
  // D = A(16x32 f16) * B(32x16 f16) + C(16x16 f32)
  return __builtin_amdgcn_wmma_f32_16x16x32_f16(
      /*neg_a=*/false, a, /*neg_b=*/false, b,
      /*c_mod=*/(short)0, c, /*reuse_a=*/false, /*reuse_b=*/false);
}

// A fragment: 16x32 f16 tile of row-major matrix (row stride = lda halfs).
// lane 0-15 : row = lane,    halves {k0+0..15}
// lane 16-31: row = lane-16, halves {k0+16..31}
__device__ __forceinline__ v16h load_a_frag(const half_t* base, int lda,
                                            int k0, int lane) {
  int row = lane & 15;
  int kl  = (lane >> 4) * 16;
  const half_t* p = base + row * lda + k0;
  v16h a;
#pragma unroll
  for (int i = 0; i < 8; ++i) a[i] = p[kl + i];
#pragma unroll
  for (int i = 0; i < 8; ++i) a[8 + i] = p[kl + 8 + i];
  return a;
}

// B fragment for Gram matmul: B[k][c] = y[m0 + c][k] (y row-major, 64 halfs).
// Each lane's 16 halves are k-contiguous -> single 32B global load.
__device__ __forceinline__ v16h load_b_frag_yt(const half_t* ybase,
                                               int m0, int k0, int lane) {
  const half_t* p = ybase + (size_t)(m0 + (lane & 15)) * CH + k0 + (lane >> 4) * 16;
  return *(const v16h*)p;
}

// Fill fragment-major B storage from a row-major f32 K x N weight matrix.
// Fragment f = kb*ntiles + nt covers k in [kb*32, kb*32+32), n in [nt*16, nt*16+16).
// dst[((f*32 + lane)*16) + i] = W[(kb*32 + (lane>>4)*16 + i)][nt*16 + (lane&15)].
__device__ __forceinline__ void fill_b_swizzled(half_t* dst, const float* W,
                                                int ldb, int kblocks, int ntiles,
                                                int tid, int nthreads) {
  const int total = kblocks * ntiles * 32 * 16;
  for (int t = tid; t < total; t += nthreads) {
    int i    = t & 15;
    int lane = (t >> 4) & 31;
    int f    = t >> 9;
    int nt   = f % ntiles;
    int kb   = f / ntiles;
    int k    = kb * 32 + (lane >> 4) * 16 + i;
    int col  = nt * 16 + (lane & 15);
    dst[t] = (half_t)W[k * ldb + col];
  }
}

__device__ __forceinline__ v16h load_b_sw(const half_t* base, int frag, int lane) {
  return *(const v16h*)(base + ((size_t)frag * 32 + lane) * 16);
}

// ---------------------------------------------------------------------------
// Kernel 1: y = x^T * W1 per (h,b); also f16 copy of y and sq = ||y||^2.
// Grid: (H*B, N/128), 256 threads (8 waves, 16 rows each).
// ---------------------------------------------------------------------------
__global__ __launch_bounds__(256) void k_conv1(const float* __restrict__ x,
                                               const float* __restrict__ W1,
                                               half_t* __restrict__ yh,
                                               float*  __restrict__ yf,
                                               float*  __restrict__ sq) {
  __shared__ half_t xs[128 * CIN];                       // xs[j][c]  (16 KB)
  __shared__ __align__(32) half_t w1sw[2 * 4 * 32 * 16]; // B frags    (8 KB)
  __shared__ float  yt[128 * CH];                        // staging   (32 KB)

  const int hb = blockIdx.x;           // h*B + b
  const int h  = hb >> 2, b = hb & 3;
  const int n0 = blockIdx.y * 128;
  const int tid = threadIdx.x;

  for (int t = tid; t < 128 * CIN; t += 256) {
    int c = t >> 7, j = t & 127;
    xs[j * CIN + c] = (half_t)x[((size_t)b * CIN + c) * NPTS + n0 + j];
  }
  fill_b_swizzled(w1sw, W1 + (size_t)h * CIN * CH, CH, 2, 4, tid, 256);
  __syncthreads();

  const int wave = tid >> 5, lane = tid & 31;
  const int r0 = wave * 16;

  v8f acc[4] = {v8f{}, v8f{}, v8f{}, v8f{}};
#pragma unroll
  for (int kb = 0; kb < 2; ++kb) {
    v16h a = load_a_frag(xs + r0 * CIN, CIN, kb * 32, lane);
#pragma unroll
    for (int nt = 0; nt < 4; ++nt)
      acc[nt] = wmma_f16(a, load_b_sw(w1sw, kb * 4 + nt, lane), acc[nt]);
  }

  // C/D layout: lane covers col = lane&15, rows = v + 8*(lane>>4)
  const int colb = lane & 15;
  const int rhi  = (lane >> 4) * 8;
#pragma unroll
  for (int nt = 0; nt < 4; ++nt)
#pragma unroll
    for (int v = 0; v < 8; ++v)
      yt[(r0 + rhi + v) * CH + nt * 16 + colb] = acc[nt][v];
  __syncthreads();

  const size_t base = ((size_t)hb * NPTS + n0) * CH;
  for (int t = tid; t < 128 * CH; t += 256) {
    float f = yt[t];
    yf[base + t] = f;
    yh[base + t] = (half_t)f;
  }
  for (int j = tid; j < 128; j += 256) {
    float s = 0.f;
#pragma unroll 8
    for (int d = 0; d < CH; ++d) { float f = yt[j * CH + d]; s += f * f; }
    sq[(size_t)hb * NPTS + n0 + j] = s;
  }
}

// ---------------------------------------------------------------------------
// Kernel 2: fused Gram-matmul + top-K.  Each block owns 16 query rows of one
// (h,b); builds dist[16][4096] in LDS via WMMA (dist never touches HBM; only
// possible because of CDNA5's 320KB WGP LDS), then 16 lanes/row perform 20
// rounds of argmax extraction (shfl width-16 reduce).
// Grid: (H*B, N/16), 256 threads.  Dynamic LDS ~258 KB.
// ---------------------------------------------------------------------------
#define DPITCH 4104   // 4096 + 8 floats pad

__global__ __launch_bounds__(256) void k_knn(const half_t* __restrict__ yh,
                                             const float*  __restrict__ sq,
                                             int* __restrict__ idx) {
  extern __shared__ char smem[];
  float*  dist  = (float*)smem;                                   // 16 x DPITCH
  half_t* atile = (half_t*)(smem + 16 * DPITCH * sizeof(float));  // 16 x 64
  float*  sqr   = (float*)(atile + 16 * CH);                      // 16

  const int hb = blockIdx.x;
  const int n0 = blockIdx.y * 16;
  const int tid = threadIdx.x;
  const half_t* ybase  = yh + (size_t)hb * NPTS * CH;
  const float*  sqbase = sq + (size_t)hb * NPTS;

  for (int t = tid; t < 16 * CH; t += 256)
    atile[t] = ybase[(size_t)n0 * CH + t];
  if (tid < 16) sqr[tid] = sqbase[n0 + tid];
  __syncthreads();

  const int wave = tid >> 5, lane = tid & 31;
  const int colb = lane & 15;
  const int rhi  = (lane >> 4) * 8;

  // A fragments shared by all column tiles -> hoist out of the loop.
  v16h a0 = load_a_frag(atile, CH, 0, lane);
  v16h a1 = load_a_frag(atile, CH, 32, lane);

  for (int mt = wave; mt < NPTS / 16; mt += 8) {   // 32 tiles/wave, 64 WMMA
    const int m0 = mt * 16;
    if (mt + 8 < NPTS / 16) {                      // prefetch next B tile
      const half_t* np = ybase + (size_t)((mt + 8) * 16 + colb) * CH
                       + (lane >> 4) * 16;
      __builtin_prefetch(np, 0, 1);
    }
    v8f acc = {};
    acc = wmma_f16(a0, load_b_frag_yt(ybase, m0, 0,  lane), acc);
    acc = wmma_f16(a1, load_b_frag_yt(ybase, m0, 32, lane), acc);
    const float sqc = sqbase[m0 + colb];
#pragma unroll
    for (int v = 0; v < 8; ++v) {
      int r = rhi + v;
      dist[r * DPITCH + m0 + colb] = 2.0f * acc[v] - sqr[r] - sqc;
    }
  }
  __syncthreads();

  // Top-K: row = tid>>4 (half-wave per row, rows stay inside one wave).
  const int row = tid >> 4;
  const int li  = tid & 15;
  float* drow = dist + row * DPITCH;
  int*   orow = idx + ((size_t)hb * NPTS + n0 + row) * KNN;

  for (int k = 0; k < KNN; ++k) {
    float bv = -3.0e38f; int bi = 0x7fffffff;
    for (int c = li; c < NPTS; c += 16) {
      float v = drow[c];
      if (v > bv || (v == bv && c < bi)) { bv = v; bi = c; }
    }
#pragma unroll
    for (int off = 8; off; off >>= 1) {
      float ov = __shfl_xor(bv, off, 16);
      int   oi = __shfl_xor(bi, off, 16);
      if (ov > bv || (ov == bv && oi < bi)) { bv = ov; bi = oi; }
    }
    if (li == 0) {
      orow[k] = bi;
      drow[bi] = -3.3e38f;   // remove winner (per-wave DS ops are in-order)
    }
  }
}

// ---------------------------------------------------------------------------
// Kernel 3: edge conv + attention.  One wave per node: gather 20 neighbors +
// center into 32x128 f16 tile (rows 20..31 zero), 32 WMMAs vs pre-swizzled
// W2 fragments, lrelu, softmax over K held in accumulator VGPRs (one lane^16
// shuffle exchange), weighted neighbor sum, lrelu + ELU, channels-first store.
// Grid: 8192 blocks x 256 threads (8 waves = 8 nodes / block).
// ---------------------------------------------------------------------------
#define ASTR  (2 * CH)                           // a-tile row stride (128 halfs)
#define PERW  (32 * ASTR * 2 + KNN * CH * 4)     // 8192 + 5120 bytes per wave

__global__ __launch_bounds__(256) void k_attn(const half_t* __restrict__ yh,
                                              const float*  __restrict__ yf,
                                              const float*  __restrict__ W2,
                                              const int*    __restrict__ idx,
                                              float* __restrict__ out) {
  extern __shared__ char smem[];
  half_t* w2sw = (half_t*)smem;                  // 4*4 B frags (16 KB)
  const int tid = threadIdx.x, wave = tid >> 5, lane = tid & 31;
  char*   wbase = smem + 4 * 4 * 32 * 16 * sizeof(half_t) + wave * PERW;
  half_t* at    = (half_t*)wbase;                    // 32 x 128 f16
  float*  nl    = (float*)(wbase + 32 * ASTR * 2);   // 20 x 64 f32 neighbors

  const int g  = blockIdx.x * 8 + wave;              // 0 .. 65535
  const int hb = g >> 12;
  const int n  = g & (NPTS - 1);
  const int h  = hb >> 2, b = hb & 3;

  fill_b_swizzled(w2sw, W2 + (size_t)h * 2 * CH * CH, CH, 4, 4, tid, 256);
  __syncthreads();

  const half_t* yhb = yh + (size_t)hb * NPTS * CH;
  const float*  yfb = yf + (size_t)hb * NPTS * CH;
  const int* myidx  = idx + ((size_t)hb * NPTS + n) * KNN;

  {
    const int r = lane;                              // row r of the A tile
    v8h* dstv = (v8h*)(at + r * ASTR);               // 16B chunks
    if (r < KNN) {
      const int nb = myidx[r];
      const v8h* s1 = (const v8h*)(yhb + (size_t)nb * CH);  // neighbor f16
      const v8h* s2 = (const v8h*)(yhb + (size_t)n  * CH);  // center f16
      const v4f* s3 = (const v4f*)(yfb + (size_t)nb * CH);  // neighbor f32
      v4f* nd = (v4f*)(nl + r * CH);
#pragma unroll
      for (int d = 0; d < 8; ++d) dstv[d]     = s1[d];
#pragma unroll
      for (int d = 0; d < 8; ++d) dstv[8 + d] = s2[d];
#pragma unroll
      for (int d = 0; d < 16; ++d) nd[d] = s3[d];
    } else {
      v8h z = {};
#pragma unroll
      for (int d = 0; d < 16; ++d) dstv[d] = z;
    }
  }
  // per-wave DS ops are in-order -> no block barrier needed before WMMA.

  v8f acc[2][4] = {{v8f{}, v8f{}, v8f{}, v8f{}}, {v8f{}, v8f{}, v8f{}, v8f{}}};
#pragma unroll
  for (int kb = 0; kb < 4; ++kb) {
    v16h a0 = load_a_frag(at,             ASTR, kb * 32, lane);
    v16h a1 = load_a_frag(at + 16 * ASTR, ASTR, kb * 32, lane);
#pragma unroll
    for (int nt = 0; nt < 4; ++nt) {
      v16h bf = load_b_sw(w2sw, kb * 4 + nt, lane);
      acc[0][nt] = wmma_f16(a0, bf, acc[0][nt]);
      acc[1][nt] = wmma_f16(a1, bf, acc[1][nt]);
    }
  }
  // leaky-relu on e = conv2 output
#pragma unroll
  for (int mt = 0; mt < 2; ++mt)
#pragma unroll
    for (int nt = 0; nt < 4; ++nt)
#pragma unroll
      for (int v = 0; v < 8; ++v) {
        float e = acc[mt][nt][v];
        acc[mt][nt][v] = e > 0.f ? e : NEG_SLOPE * e;
      }

  // Softmax over K rows (per column), weighted sum, lrelu + ELU, store.
  // Lane owns col = nt*16 + (lane&15); rows held: hi==0 -> k in {0..7,16..19},
  // hi==1 -> k in {8..15}; partner lane is lane^16.
  const int colb = lane & 15;
  const int hi   = lane >> 4;
#pragma unroll
  for (int nt = 0; nt < 4; ++nt) {
    const int col = nt * 16 + colb;
    float m = -3.0e38f;
#pragma unroll
    for (int v = 0; v < 8; ++v) m = fmaxf(m, acc[0][nt][v]);
    if (hi == 0) {
#pragma unroll
      for (int v = 0; v < 4; ++v) m = fmaxf(m, acc[1][nt][v]);
    }
    m = fmaxf(m, __shfl_xor(m, 16));

    float s = 0.f, d = 0.f;
#pragma unroll
    for (int v = 0; v < 8; ++v) {
      int k = 8 * hi + v;
      float ev = __expf(acc[0][nt][v] - m);
      s += ev; d += ev * nl[k * CH + col];
    }
    if (hi == 0) {
#pragma unroll
      for (int v = 0; v < 4; ++v) {
        int k = 16 + v;
        float ev = __expf(acc[1][nt][v] - m);
        s += ev; d += ev * nl[k * CH + col];
      }
    }
    s += __shfl_xor(s, 16);
    d += __shfl_xor(d, 16);

    float o = d / s;
    o = o > 0.f ? o : NEG_SLOPE * o;            // per-head lrelu
    o = o > 0.f ? o : (__expf(o) - 1.0f);       // ELU after concat
    if (hi == 0)
      out[((size_t)b * (HEADS * CH) + h * CH + col) * NPTS + n] = o;
  }
}

// ---------------------------------------------------------------------------
// Host launcher
// ---------------------------------------------------------------------------
static constexpr size_t YH_OFF  = 0;                                   // f16 y
static constexpr size_t YF_OFF  = YH_OFF + (size_t)HEADS * BATCH * NPTS * CH * 2;
static constexpr size_t SQ_OFF  = YF_OFF + (size_t)HEADS * BATCH * NPTS * CH * 4;
static constexpr size_t IDX_OFF = SQ_OFF + (size_t)HEADS * BATCH * NPTS * 4;

static constexpr int KNN_SMEM  = 16 * DPITCH * 4 + 16 * CH * 2 + 16 * 4;
static constexpr int ATTN_SMEM = 4 * 4 * 32 * 16 * 2 + 8 * PERW;

extern "C" void kernel_launch(void* const* d_in, const int* in_sizes, int n_in,
                              void* d_out, int out_size, void* d_ws, size_t ws_size,
                              hipStream_t stream) {
  (void)in_sizes; (void)n_in; (void)out_size; (void)ws_size;
  const float* x  = (const float*)d_in[0];
  const float* W1 = (const float*)d_in[1];
  const float* W2 = (const float*)d_in[2];
  float* out = (float*)d_out;

  char* ws = (char*)d_ws;
  half_t* yh  = (half_t*)(ws + YH_OFF);
  float*  yf  = (float*)(ws + YF_OFF);
  float*  sq  = (float*)(ws + SQ_OFF);
  int*    idx = (int*)(ws + IDX_OFF);

  (void)hipFuncSetAttribute((const void*)k_knn,
      hipFuncAttributeMaxDynamicSharedMemorySize, KNN_SMEM);
  (void)hipFuncSetAttribute((const void*)k_attn,
      hipFuncAttributeMaxDynamicSharedMemorySize, ATTN_SMEM);

  k_conv1<<<dim3(HEADS * BATCH, NPTS / 128), 256, 0, stream>>>(x, W1, yh, yf, sq);
  k_knn  <<<dim3(HEADS * BATCH, NPTS / 16),  256, KNN_SMEM,  stream>>>(yh, sq, idx);
  k_attn <<<dim3((HEADS * BATCH * NPTS) / 8), 256, ATTN_SMEM, stream>>>(yh, yf, W2, idx, out);
}
